// DistSAGEConv_59382217834733
// MI455X (gfx1250) — matrix operand order repaired and checked by reading, not compiled, not run on hardware
//
#include <hip/hip_runtime.h>

#define N_NODES 50000
#define N_EDGES 800000
#define D       512
#define M_TILES (N_NODES / 16)   // 3125
#define N_TILES (D / 16)         // 32

typedef __attribute__((ext_vector_type(16))) __bf16 v16bf;
typedef __attribute__((ext_vector_type(8)))  __bf16 v8bf;
typedef __attribute__((ext_vector_type(8)))  float  v8f;
typedef __attribute__((ext_vector_type(4)))  float  v4f;

// ---------------- f32 -> bf16 conversion (8 elements / thread) ----------------
__global__ __launch_bounds__(256) void cvt_f32_bf16(const float* __restrict__ src,
                                                    __bf16* __restrict__ dst, int n8) {
    int i = blockIdx.x * 256 + threadIdx.x;
    if (i >= n8) return;
    const float* p = src + (long)i * 8;
    v4f lo = *(const v4f*)(p);
    v4f hi = *(const v4f*)(p + 4);
    v8bf o;
#pragma unroll
    for (int j = 0; j < 4; ++j) { o[j] = (__bf16)lo[j]; o[j + 4] = (__bf16)hi[j]; }
    *(v8bf*)(dst + (long)i * 8) = o;
}

// ---------------- CSR build: histogram of dst ----------------
__global__ __launch_bounds__(256) void hist_kernel(const int* __restrict__ dst,
                                                   int* __restrict__ cnt) {
    int e = blockIdx.x * 256 + threadIdx.x;
    if (e < N_EDGES) atomicAdd(&cnt[dst[e]], 1);
}

// ---------------- CSR build: single-workgroup exclusive scan (LDS) ----------------
__global__ __launch_bounds__(256) void scan_kernel(const int* __restrict__ cnt,
                                                   int* __restrict__ row_start,
                                                   int* __restrict__ cursor) {
    __shared__ int part[256];
    const int t = threadIdx.x;
    const int CH = (N_NODES + 255) / 256;           // 196
    const int lo = t * CH;
    const int hi = (lo + CH < N_NODES) ? lo + CH : N_NODES;
    int s = 0;
    for (int i = lo; i < hi; ++i) s += cnt[i];
    part[t] = s;
    __syncthreads();
    // Hillis-Steele inclusive scan over 256 partials
    for (int off = 1; off < 256; off <<= 1) {
        int v = part[t];
        int u = (t >= off) ? part[t - off] : 0;
        __syncthreads();
        part[t] = v + u;
        __syncthreads();
    }
    int run = (t == 0) ? 0 : part[t - 1];
    for (int i = lo; i < hi; ++i) {
        int c = cnt[i];
        row_start[i] = run;
        cursor[i]    = run;
        run += c;
    }
    if (t == 255) row_start[N_NODES] = run;         // == N_EDGES
}

// ---------------- CSR build: fill column list via cursor ----------------
__global__ __launch_bounds__(256) void fill_kernel(const int* __restrict__ src,
                                                   const int* __restrict__ dst,
                                                   int* __restrict__ cursor,
                                                   int* __restrict__ col) {
    int e = blockIdx.x * 256 + threadIdx.x;
    if (e < N_EDGES) {
        int p = atomicAdd(&cursor[dst[e]], 1);
        col[p] = src[e];
    }
}

// ---------------- WMMA GEMM core: Z = X (bf16) @ W^T (bf16), 16x16 tile per wave ----------------
// A-frag: lane m = lane&15, K halves split by lane>>4 per ISA layout.
// B-frag: lane n = lane&15, same K split by lane>>4 as A (row of W = column of B).
__device__ __forceinline__ v8f wmma_tile(const __bf16* __restrict__ X,
                                         const __bf16* __restrict__ W,
                                         int mt, int nt, int lane) {
    const int r16  = lane & 15;
    const int half = lane >> 4;
    const __bf16* ap = X + (long)(mt * 16 + r16) * D;   // A row m
    const __bf16* bp = W + (long)(nt * 16 + r16) * D;   // W row n (== B column n)
    v8f acc = {};
#pragma unroll
    for (int k0 = 0; k0 < D; k0 += 32) {
        v8bf alo = *(const v8bf*)(ap + k0 + half * 8);        // K = k0 + half*8 .. +8
        v8bf ahi = *(const v8bf*)(ap + k0 + 16 + half * 8);   // K = k0+16+half*8 .. +8
        v16bf a = __builtin_shufflevector(alo, ahi, 0, 1, 2, 3, 4, 5, 6, 7,
                                          8, 9, 10, 11, 12, 13, 14, 15);
        v8bf blo = *(const v8bf*)(bp + k0 + half * 8);        // K = k0 + half*8 .. +8
        v8bf bhi = *(const v8bf*)(bp + k0 + 16 + half * 8);   // K = k0+16+half*8 .. +8
        v16bf b = __builtin_shufflevector(blo, bhi, 0, 1, 2, 3, 4, 5, 6, 7,
                                          8, 9, 10, 11, 12, 13, 14, 15);
        acc = __builtin_amdgcn_wmma_f32_16x16x32_bf16(
            false, a, false, b, (short)0, acc, false, false);
    }
    return acc;
}

__global__ __launch_bounds__(256) void gemm_neigh_kernel(const __bf16* __restrict__ X,
                                                         const __bf16* __restrict__ W,
                                                         __bf16* __restrict__ Z) {
    const int wave = (blockIdx.x << 3) + (threadIdx.x >> 5);
    const int lane = threadIdx.x & 31;
    const int mt = wave >> 5;       // / N_TILES
    const int nt = wave & 31;
    v8f acc = wmma_tile(X, W, mt, nt, lane);
    const int ncol = nt * 16 + (lane & 15);
    const int half = lane >> 4;
#pragma unroll
    for (int r = 0; r < 8; ++r) {
        int m = mt * 16 + half * 8 + r;                 // D layout: VGPR r -> M = 8*half + r
        Z[(long)m * D + ncol] = (__bf16)acc[r];
    }
}

// ---------------- CSR gather: out[n,:] = (sum_j Z[col[j],:]) / max(deg,1), wave per node ----------
__global__ __launch_bounds__(256) void gather_kernel(const __bf16* __restrict__ Z,
                                                     const int* __restrict__ row_start,
                                                     const int* __restrict__ col,
                                                     float* __restrict__ out) {
    const int wave = (blockIdx.x << 3) + (threadIdx.x >> 5);
    const int lane = threadIdx.x & 31;
    if (wave >= N_NODES) return;
    const int start = row_start[wave];
    const int end   = row_start[wave + 1];
    v8f a0 = {}, a1 = {};
    for (int j = start; j < end; ++j) {
        int s = col[j];
        if (j + 1 < end)   // hide dependent col->row latency with a gfx1250 prefetch
            __builtin_prefetch(Z + (long)col[j + 1] * D + lane * 16, 0, 3);
        const __bf16* zp = Z + (long)s * D + lane * 16;
        v8bf z0 = *(const v8bf*)(zp);
        v8bf z1 = *(const v8bf*)(zp + 8);
#pragma unroll
        for (int i = 0; i < 8; ++i) { a0[i] += (float)z0[i]; a1[i] += (float)z1[i]; }
    }
    const int degi = end - start;
    const float r = 1.0f / (float)(degi > 1 ? degi : 1);
    float* op = out + (long)wave * D + lane * 16;
#pragma unroll
    for (int i = 0; i < 8; ++i) { a0[i] *= r; a1[i] *= r; }
    *(v8f*)(op)     = a0;
    *(v8f*)(op + 8) = a1;
}

// ---------------- fused finish: out = relu(X @ Wself^T + out + b), in place ----------------
__global__ __launch_bounds__(256) void finish_kernel(const __bf16* __restrict__ X,
                                                     const __bf16* __restrict__ W,
                                                     const float* __restrict__ bias,
                                                     float* __restrict__ out) {
    const int wave = (blockIdx.x << 3) + (threadIdx.x >> 5);
    const int lane = threadIdx.x & 31;
    const int mt = wave >> 5;
    const int nt = wave & 31;
    v8f acc = wmma_tile(X, W, mt, nt, lane);
    const int ncol = nt * 16 + (lane & 15);
    const int half = lane >> 4;
    const float bn = bias[ncol];
#pragma unroll
    for (int r = 0; r < 8; ++r) {
        int m = mt * 16 + half * 8 + r;
        long idx = (long)m * D + ncol;
        float v = acc[r] + out[idx] + bn;   // read h_neigh, then overwrite
        out[idx] = fmaxf(v, 0.0f);
    }
}

// ---------------- host launch ----------------
extern "C" void kernel_launch(void* const* d_in, const int* in_sizes, int n_in,
                              void* d_out, int out_size, void* d_ws, size_t ws_size,
                              hipStream_t stream) {
    const float* X      = (const float*)d_in[0];
    const int*   src    = (const int*)d_in[1];
    const int*   dst    = (const int*)d_in[2];
    // d_in[3] = layer (hop selector; aggregation math identical -> unused)
    const float* Wself  = (const float*)d_in[4];
    const float* Wneigh = (const float*)d_in[5];
    const float* bias   = (const float*)d_in[6];
    float* out = (float*)d_out;

    // workspace layout (512B-aligned offsets), ~107 MB total
    char* ws = (char*)d_ws;
    int*    cnt       = (int*)(ws + 0);             //   200,704 B
    int*    row_start = (int*)(ws + 200704);        //   200,704 B (N+1 ints)
    int*    cursor    = (int*)(ws + 401408);        //   200,704 B
    int*    col       = (int*)(ws + 602112);        // 3,200,512 B
    __bf16* Xb        = (__bf16*)(ws + 3802624);    // 51,200,000 B
    __bf16* Wsb       = (__bf16*)(ws + 55002624);   //    524,288 B
    __bf16* Wnb       = (__bf16*)(ws + 55526912);   //    524,288 B
    __bf16* Zn        = (__bf16*)(ws + 56051200);   // 51,200,000 B

    hipMemsetAsync(cnt, 0, (size_t)N_NODES * sizeof(int), stream);

    const int xn8 = N_NODES * D / 8;
    const int wn8 = D * D / 8;
    cvt_f32_bf16<<<(xn8 + 255) / 256, 256, 0, stream>>>(X, Xb, xn8);
    cvt_f32_bf16<<<(wn8 + 255) / 256, 256, 0, stream>>>(Wself, Wsb, wn8);
    cvt_f32_bf16<<<(wn8 + 255) / 256, 256, 0, stream>>>(Wneigh, Wnb, wn8);

    // CSR build
    hist_kernel<<<(N_EDGES + 255) / 256, 256, 0, stream>>>(dst, cnt);
    scan_kernel<<<1, 256, 0, stream>>>(cnt, row_start, cursor);
    fill_kernel<<<(N_EDGES + 255) / 256, 256, 0, stream>>>(src, dst, cursor, col);

    // neighbor projection (WMMA), then CSR gather of projected rows
    gemm_neigh_kernel<<<M_TILES * N_TILES / 8, 256, 0, stream>>>(Xb, Wnb, Zn);
    gather_kernel<<<(N_NODES + 7) / 8, 256, 0, stream>>>(Zn, row_start, col, out);

    // self projection (WMMA) + bias + ReLU, fused in place
    finish_kernel<<<M_TILES * N_TILES / 8, 256, 0, stream>>>(Xb, Wsb, bias, out);
}
